// QuantumKernelMethod_65481071401444
// MI455X (gfx1250) — matrix operand order repaired and checked by reading, not compiled, not run on hardware
//
#include <hip/hip_runtime.h>
#include <stdint.h>

typedef __bf16   v16bf __attribute__((ext_vector_type(16)));
typedef float    v8f   __attribute__((ext_vector_type(8)));
typedef float    f32x4 __attribute__((ext_vector_type(4)));
typedef uint32_t u32x4 __attribute__((ext_vector_type(4)));
typedef uint32_t u32x8 __attribute__((ext_vector_type(8)));

namespace {
constexpr int kN = 8192;   // rows of x
constexpr int kM = 8192;   // rows of y
constexpr int kD = 512;    // feature dim

constexpr int BM = 128;    // output tile rows per workgroup
constexpr int BN = 128;    // output tile cols per workgroup
constexpr int BK = 32;     // bf16 K-chunk per stage (one 16x16x32 WMMA k-step)

constexpr int LDS_ROW = 80;               // 32 bf16 = 64B data + 16B pad (TDM-inserted)
constexpr int TILE_B  = 128 * LDS_ROW;    // 10240 B per staged tile
constexpr int BUF_B   = 2 * TILE_B;       // A tile + B tile per stage = 20480 B
constexpr int NSTAGE  = 3;                // triple-buffered TDM pipeline (61440 B LDS)
constexpr int NCHUNK  = kD / BK;          // 16 stages
}

__device__ __forceinline__ uint16_t f32_to_bf16_rne(float f) {
  uint32_t u = __float_as_uint(f);
  u += 0x7FFFu + ((u >> 16) & 1u);
  return (uint16_t)(u >> 16);
}

// ---------------------------------------------------------------------------
// Prep: fp32 -> bf16 copies of x and y into workspace, plus row squared-norms.
// One wave (32 lanes) per row of 512 floats. fp32 source is read-once -> NT
// loads; bf16 output is about to be re-read by the GEMM -> regular (RT) stores
// so it stays resident in the 192MB L2.
// ---------------------------------------------------------------------------
__global__ __launch_bounds__(256) void prep_kernel(
    const float* __restrict__ x, const float* __restrict__ y,
    uint16_t* __restrict__ xb, uint16_t* __restrict__ yb,
    float* __restrict__ xsq, float* __restrict__ ysq) {
  const int wave = (int)((blockIdx.x * blockDim.x + threadIdx.x) >> 5);
  const int lane = (int)(threadIdx.x & 31);

  const float* src;
  uint16_t* dst;
  float* nrm;
  int row;
  if (wave < kN) { src = x; dst = xb; nrm = xsq; row = wave; }
  else           { src = y; dst = yb; nrm = ysq; row = wave - kN; }

  const f32x4* s = (const f32x4*)(src + (size_t)row * kD);
  uint16_t* d = dst + (size_t)row * kD;

  float acc = 0.0f;
#pragma unroll
  for (int i = 0; i < kD / 4 / 32; ++i) {            // 4 iterations
    f32x4 v = __builtin_nontemporal_load(&s[lane + 32 * i]);
    acc += v.x * v.x + v.y * v.y + v.z * v.z + v.w * v.w;
    uint32_t p0 = (uint32_t)f32_to_bf16_rne(v.x) | ((uint32_t)f32_to_bf16_rne(v.y) << 16);
    uint32_t p1 = (uint32_t)f32_to_bf16_rne(v.z) | ((uint32_t)f32_to_bf16_rne(v.w) << 16);
    *(uint2*)(d + 4 * (lane + 32 * i)) = make_uint2(p0, p1);
  }
#pragma unroll
  for (int off = 16; off > 0; off >>= 1) acc += __shfl_xor(acc, off, 32);
  if (lane == 0) nrm[row] = acc;
}

// ---------------------------------------------------------------------------
// TDM: issue one tensor_load_to_lds for a 128-row x 32-col bf16 tile.
// D# group1: data_size=2B, pad_enable, pad every 16 dwords (64B row) by
// 4 dwords (16B) -> LDS row stride 80B (16B aligned, bank-conflict-free).
// tensor_dim = tile_dim (tiles always fully in-bounds), memory row stride 512.
// ---------------------------------------------------------------------------
__device__ __forceinline__ void tdm_load_tile(uint32_t lds_addr, const uint16_t* gsrc) {
  uint64_t ga = (uint64_t)(uintptr_t)gsrc;
  // Force wave-uniform SGPR-resident descriptor dwords.
  uint32_t w0 = (uint32_t)__builtin_amdgcn_readfirstlane(1);                         // count=1
  uint32_t w1 = (uint32_t)__builtin_amdgcn_readfirstlane((int)lds_addr);             // lds_addr
  uint32_t w2 = (uint32_t)__builtin_amdgcn_readfirstlane((int)(uint32_t)ga);         // gaddr lo
  uint32_t w3 = (uint32_t)__builtin_amdgcn_readfirstlane(
      (int)(((uint32_t)(ga >> 32) & 0x01FFFFFFu) | 0x80000000u));                    // gaddr hi | type=2
  u32x4 g0 = {w0, w1, w2, w3};
  u32x8 g1 = {
      0x06D10000u,     // data_size=1(2B) | pad_enable | pad_interval=3(16dw) | pad_amount=3(4dw)
      32u << 16,       // tensor_dim0 = 32
      128u << 16,      // tensor_dim1 = 128
      32u << 16,       // tile_dim0 = 32
      128u,            // tile_dim1 = 128, tile_dim2 = 0
      512u,            // tensor_dim0_stride = 512 (lo 32)
      0u, 0u};         // stride hi / dim1_stride = 0
  asm volatile("tensor_load_to_lds %0, %1" :: "s"(g0), "s"(g1) : "memory");
}

// ---------------------------------------------------------------------------
// Fused GEMM (bf16 WMMA, fp32 accum) + RBF epilogue.
// Workgroup: 128x128 tile, 8 waves, each wave 32(M) x 64(N) = 2x4 WMMA tiles.
// Triple-buffered TDM staging, wave 0 is the DMA issuer (2 stages ahead).
// ---------------------------------------------------------------------------
__global__ __launch_bounds__(256) void rbf_wmma_kernel(
    const uint16_t* __restrict__ xb, const uint16_t* __restrict__ yb,
    const float* __restrict__ xsq, const float* __restrict__ ysq,
    const float* __restrict__ gamma_p, float* __restrict__ out) {
  __shared__ __align__(16) unsigned char sm[NSTAGE * BUF_B];   // 61440 B

  const int tid  = (int)threadIdx.x;
  const int lane = tid & 31;
  const int wv   = tid >> 5;            // wave 0..7
  const int gM0  = (int)blockIdx.y * BM;
  const int gN0  = (int)blockIdx.x * BN;
  const int wm   = (wv >> 1) * 32;      // wave M offset in tile
  const int wn   = (wv & 1) * 64;       // wave N offset in tile

  v8f acc[2][4];
#pragma unroll
  for (int m = 0; m < 2; ++m)
#pragma unroll
    for (int n = 0; n < 4; ++n) acc[m][n] = (v8f){0.f,0.f,0.f,0.f,0.f,0.f,0.f,0.f};

  const uint32_t lds_base = (uint32_t)(uintptr_t)(void*)sm;
  const uint16_t* aRow = xb + (size_t)gM0 * kD;   // 128 rows of x
  const uint16_t* bRow = yb + (size_t)gN0 * kD;   // 128 rows of y

  const bool issuer = (tid < 32);
  if (issuer) {                                   // stages 0,1 -> buffers 0,1
    tdm_load_tile(lds_base,                  aRow);
    tdm_load_tile(lds_base + TILE_B,         bRow);
    tdm_load_tile(lds_base + BUF_B,          aRow + BK);
    tdm_load_tile(lds_base + BUF_B + TILE_B, bRow + BK);
  }

  const int sel = lane & 15;
  const int grp = lane >> 4;

#pragma unroll
  for (int kc = 0; kc < NCHUNK; ++kc) {
    const int cur = kc % NSTAGE;
    // Top barrier: all waves done reading buf[(kc-1)%3] == buf[(kc+2)%3],
    // which is exactly the buffer the issuer is about to overwrite.
    __syncthreads();
    if (issuer) {
      if (kc + 2 < NCHUNK) {
        const uint32_t nb = lds_base + (uint32_t)(((kc + 2) % NSTAGE) * BUF_B);
        tdm_load_tile(nb,          aRow + (size_t)(kc + 2) * BK);
        tdm_load_tile(nb + TILE_B, bRow + (size_t)(kc + 2) * BK);
        __builtin_amdgcn_s_wait_tensorcnt(4);   // stages kc+1,kc+2 in flight; kc done
      } else if (kc + 1 < NCHUNK) {
        __builtin_amdgcn_s_wait_tensorcnt(2);   // stage kc+1 in flight; kc done
      } else {
        __builtin_amdgcn_s_wait_tensorcnt(0);   // last stage done
      }
    }
    __syncthreads();                            // buf[cur] visible to all waves

    const unsigned aBase = (unsigned)(cur * BUF_B);
    const unsigned bBase = aBase + TILE_B;

    union Frag { uint4 q[2]; v16bf v; };
    Frag a[2], b[4];
    // A fragment (16x32 bf16): lane-group 0 -> K 0..7 / 16..23, group 1 -> K 8..15 / 24..31
#pragma unroll
    for (int mt = 0; mt < 2; ++mt) {
      const unsigned o = aBase + (unsigned)(wm + mt * 16 + sel) * LDS_ROW + (unsigned)(grp * 16);
      a[mt].q[0] = *(const uint4*)(sm + o);
      a[mt].q[1] = *(const uint4*)(sm + o + 32);
    }
    // B fragment (32x16 bf16): lane-group 0 -> K 0..7 / 16..23, group 1 -> K 8..15 / 24..31
#pragma unroll
    for (int nt = 0; nt < 4; ++nt) {
      const unsigned o = bBase + (unsigned)(wn + nt * 16 + sel) * LDS_ROW + (unsigned)(grp * 16);
      b[nt].q[0] = *(const uint4*)(sm + o);
      b[nt].q[1] = *(const uint4*)(sm + o + 32);
    }
#pragma unroll
    for (int mt = 0; mt < 2; ++mt)
#pragma unroll
      for (int nt = 0; nt < 4; ++nt)
        acc[mt][nt] = __builtin_amdgcn_wmma_f32_16x16x32_bf16(
            false, a[mt].v, false, b[nt].v, (short)0, acc[mt][nt], false, false);
  }

  // Epilogue: d2 = ||x||^2 + ||y||^2 - 2*cross, out = exp(-gamma*max(d2,0)).
  // Output is write-once streaming (256MB) -> non-temporal stores keep the
  // bf16 operand copies resident in L2.
  const float g = *gamma_p;
  float xs[2][8];
#pragma unroll
  for (int mt = 0; mt < 2; ++mt)
#pragma unroll
    for (int i = 0; i < 8; ++i)
      xs[mt][i] = xsq[gM0 + wm + mt * 16 + grp * 8 + i];

#pragma unroll
  for (int nt = 0; nt < 4; ++nt) {
    const int gN = gN0 + wn + nt * 16 + sel;
    const float yn = ysq[gN];
#pragma unroll
    for (int mt = 0; mt < 2; ++mt) {
#pragma unroll
      for (int i = 0; i < 8; ++i) {
        const int gM = gM0 + wm + mt * 16 + grp * 8 + i;
        float d2 = xs[mt][i] + yn - 2.0f * acc[mt][nt][i];
        d2 = fmaxf(d2, 0.0f);
        __builtin_nontemporal_store(__expf(-g * d2), &out[(size_t)gM * kM + gN]);
      }
    }
  }
}

// ---------------------------------------------------------------------------
// Workspace layout (requires ws_size >= 16,842,752 bytes):
//   [0, 8MB)        xb  bf16[8192][512]
//   [8MB, 16MB)     yb  bf16[8192][512]
//   [16MB, +32KB)   xsq float[8192]
//   [.., +32KB)     ysq float[8192]
// ---------------------------------------------------------------------------
extern "C" void kernel_launch(void* const* d_in, const int* in_sizes, int n_in,
                              void* d_out, int out_size, void* d_ws, size_t ws_size,
                              hipStream_t stream) {
  (void)in_sizes; (void)n_in; (void)out_size; (void)ws_size;
  const float* x     = (const float*)d_in[0];
  const float* y     = (const float*)d_in[1];
  const float* gamma = (const float*)d_in[2];
  float* out = (float*)d_out;

  char* ws = (char*)d_ws;
  uint16_t* xbuf = (uint16_t*)ws;
  uint16_t* ybuf = (uint16_t*)(ws + (size_t)kN * kD * sizeof(uint16_t));
  float* xsq = (float*)(ws + (size_t)(kN + kM) * kD * sizeof(uint16_t));
  float* ysq = xsq + kN;

  prep_kernel<<<(kN + kM) / 8, 256, 0, stream>>>(x, y, xbuf, ybuf, xsq, ysq);

  dim3 grid(kM / BN, kN / BM);   // (64, 64)
  rbf_wmma_kernel<<<grid, 256, 0, stream>>>(xbuf, ybuf, xsq, ysq, gamma, out);
}
